// STUNative_41601053229845
// MI455X (gfx1250) — matrix operand order repaired and checked by reading, not compiled, not run on hardware
//
#include <hip/hip_runtime.h>
#include <math.h>

// ---- model config (matches reference) ----
#define D_IN   32
#define NCH    128
#define PTRN   2000
#define KQ     48
#define NLAY   2
#define RFILT  2
#define TLEN   2048
#define HID    512
#define BATCH  8
#define MROWS  (BATCH * TLEN)   // 16384
#define EPSV   1e-6f

typedef _Float16 v16h __attribute__((ext_vector_type(16)));
typedef _Float16 v8h  __attribute__((ext_vector_type(8)));
typedef _Float16 v4h  __attribute__((ext_vector_type(4)));
typedef float    v8f  __attribute__((ext_vector_type(8)));

typedef unsigned int u32x4 __attribute__((ext_vector_type(4)));
typedef int          i32x4 __attribute__((ext_vector_type(4)));
typedef int          i32x8 __attribute__((ext_vector_type(8)));

enum { MODE_F16 = 0, MODE_GELU = 1, MODE_ADDF32 = 2, MODE_ZT = 3 };

#if __has_builtin(__builtin_amdgcn_tensor_load_to_lds) && __has_builtin(__builtin_amdgcn_s_wait_tensorcnt)
#define USE_TDM 1
#else
#define USE_TDM 0
#endif

__device__ __forceinline__ float gelu_exact(float x) {
    return 0.5f * x * (1.0f + erff(x * 0.70710678118654752f));
}

#if USE_TDM
// Build a 2D Tensor-Data-Mover descriptor (CDNA5 ISA)
// and DMA a tile of `rows` rows x `row_units` 8-byte units from global into LDS.
// data_size=3 (8B units); tensor dims set huge so the in-bounds tile never clips.
__device__ __forceinline__ void tdm_load_tile(const void* gaddr, unsigned lds_byte_off,
                                              unsigned row_units, unsigned rows,
                                              unsigned stride_units) {
    unsigned long long ga = (unsigned long long)(size_t)gaddr;
    const unsigned TD0 = 0x7fffffffu, TD1 = 0x7fffffffu;
    u32x4 g0;
    g0[0] = 1u;                                   // count=1 (valid), is_restore=0, gather off
    g0[1] = lds_byte_off;                         // lds_addr
    g0[2] = (unsigned)ga;                         // global_addr[31:0]
    g0[3] = (unsigned)((ga >> 32) & 0x01ffffffu)  // global_addr[56:32]
          | (2u << 30);                           // type=2 ("image")
    i32x8 g1;
    g1[0] = (int)(3u << 16);                      // workgroup_mask=0, data_size=3 (8B)
    g1[1] = (int)((TD0 & 0xffffu) << 16);         // atomic_barrier_addr=0 | tensor_dim0 lo
    g1[2] = (int)(((TD0 >> 16) & 0xffffu) | ((TD1 & 0xffffu) << 16));
    g1[3] = (int)(((TD1 >> 16) & 0xffffu) | (row_units << 16));   // tile_dim0
    g1[4] = (int)(rows & 0xffffu);                // tile_dim1 (tile_dim2=0)
    g1[5] = (int)stride_units;                    // tensor_dim0_stride[31:0]
    g1[6] = 0;                                    // stride0[47:32]=0, stride1 lo=0
    g1[7] = 0;                                    // stride1 hi=0
    i32x4 gz = {0, 0, 0, 0};
#if defined(__clang_major__) && (__clang_major__ >= 23)
    i32x8 gz8 = {0, 0, 0, 0, 0, 0, 0, 0};
    __builtin_amdgcn_tensor_load_to_lds(g0, g1, gz, gz, gz8, 0);
#else
    __builtin_amdgcn_tensor_load_to_lds(g0, g1, gz, gz, 0);
#endif
}
#endif

// ---------------- weight prep kernels ----------------

__global__ void k_f2h(const float* __restrict__ a, _Float16* __restrict__ o, int n) {
    for (int i = blockIdx.x * blockDim.x + threadIdx.x; i < n; i += gridDim.x * blockDim.x)
        o[i] = (_Float16)a[i];
}

// dst[l][o][k*NCH+c] = full_mix[l][k][c][o]
__global__ void k_mixT(const float* __restrict__ mix, _Float16* __restrict__ o, int total) {
    for (int i = blockIdx.x * blockDim.x + threadIdx.x; i < total; i += gridDim.x * blockDim.x) {
        int kc = i & (RFILT * NCH - 1);
        int oc = (i >> 8) & (NCH - 1);
        int l  = i >> 15;
        int k = kc >> 7, c = kc & (NCH - 1);
        o[i] = (_Float16)mix[(((size_t)l * RFILT + k) * NCH + c) * NCH + oc];
    }
}

// toep[r][t][s] = (s<=t) ? phi[t-s][r] : 0   ; phi is (T, R) row-major
__global__ void k_toep(const float* __restrict__ phi, _Float16* __restrict__ toep, int total) {
    for (int i = blockIdx.x * blockDim.x + threadIdx.x; i < total; i += gridDim.x * blockDim.x) {
        int s = i & (TLEN - 1);
        int t = (i >> 11) & (TLEN - 1);
        int r = i >> 22;
        toep[i] = (s <= t) ? (_Float16)phi[(size_t)(t - s) * RFILT + r] : (_Float16)0.0f;
    }
}

// ---------------- embedding ----------------
__global__ void k_embed(const float* __restrict__ Xtr, const float* __restrict__ ytr,
                        const float* __restrict__ Xq, const float* __restrict__ Win,
                        float* __restrict__ h, int total) {
    for (int i = blockIdx.x * blockDim.x + threadIdx.x; i < total; i += gridDim.x * blockDim.x) {
        int n = i & (NCH - 1);
        int t = (i >> 7) & (TLEN - 1);
        int b = i >> 18;
        const float* w = Win + (size_t)n * (D_IN + 1);
        float acc = 0.0f;
        if (t < PTRN) {
            #pragma unroll 8
            for (int f = 0; f < D_IN; ++f)
                acc += Xtr[((size_t)b * D_IN + f) * PTRN + t] * w[f];
            acc += ytr[(size_t)b * PTRN + t] * w[D_IN];
        } else {
            int q = t - PTRN;
            #pragma unroll 8
            for (int f = 0; f < D_IN; ++f)
                acc += Xq[((size_t)b * D_IN + f) * KQ + q] * w[f];
        }
        h[i] = acc;
    }
}

// ---------------- rmsnorm -> f16 (one wave32 per 128-wide row) ----------------
__global__ __launch_bounds__(128) void k_rmsnorm(const float* __restrict__ h,
                                                 _Float16* __restrict__ o) {
    int row  = blockIdx.x * 4 + (threadIdx.x >> 5);
    int lane = threadIdx.x & 31;
    const float4 x = *(const float4*)(h + (size_t)row * NCH + lane * 4);
    float ss = x.x * x.x + x.y * x.y + x.z * x.z + x.w * x.w;
    #pragma unroll
    for (int off = 16; off; off >>= 1) ss += __shfl_xor(ss, off, 32);
    float s = rsqrtf(ss * (1.0f / NCH) + EPSV);
    v4h r;
    r[0] = (_Float16)(x.x * s); r[1] = (_Float16)(x.y * s);
    r[2] = (_Float16)(x.z * s); r[3] = (_Float16)(x.w * s);
    *(v4h*)(o + (size_t)row * NCH + lane * 4) = r;
}

// ---------------- generic WMMA GEMM: Out = A(MxK,f16) @ Bw(Nout x K, f16)^T ----------------
// 4 waves / block; block = one 16-row M-tile x 64 output cols.
// A tile (16x64) staged into LDS by the Tensor Data Mover (wave 0 issues the DMA,
// waits TENSORcnt, workgroup barrier releases readers); 2 WMMAs per stage.
// Batched via grid.z: b=z/nf, f=z%nf; causal clamps the K loop at the Toeplitz diagonal.
__global__ __launch_bounds__(128) void k_gemm(
    const _Float16* __restrict__ Aall, const _Float16* __restrict__ Ball,
    void* __restrict__ Out,
    int Nout, int Kdim,
    long long sA_f, long long sB_b, long long sO_b, long long sO_f,
    int nf, int ld_out, int mode, int causal, int Tdim)
{
    __shared__ _Float16 sA[16][64];   // 16x64 f16 tile (2 KB), rows contiguous (TDM layout)

    const int z  = blockIdx.z;
    const int bb = z / nf, ff = z % nf;
    const _Float16* A  = Aall + (long long)ff * sA_f;
    const _Float16* Bw = Ball + (long long)bb * sB_b;

    const int m0    = blockIdx.x * 16;
    const int wave  = threadIdx.x >> 5;
    const int lane  = threadIdx.x & 31;
    const int ntile = blockIdx.y * 64 + wave * 16;

    const int kmax   = causal ? ((Kdim < m0 + 16) ? Kdim : (m0 + 16)) : Kdim;
    const int ksteps = (kmax + 63) >> 6;

    v8f acc;
    #pragma unroll
    for (int i = 0; i < 8; ++i) acc[i] = 0.0f;

    const int fr = lane & 15;               // A-frag row
    const int fk = (lane >> 4) * 16;        // A-frag K sub-offset
    const int bn = ntile + (lane & 15);     // B-frag column (output col)
    const int bk = (lane >> 4) * 16;        // B-frag K sub-offset

#if USE_TDM
    const unsigned ldsOff = (unsigned)(size_t)(void*)&sA[0][0];
#else
    const int arow = threadIdx.x >> 3;        // 0..15
    const int acol = (threadIdx.x & 7) * 8;   // 0,8,...,56
#endif

    for (int ks = 0; ks < ksteps; ++ks) {
        const int k0 = ks << 6;
#if USE_TDM
        if (wave == 0) {
            tdm_load_tile(A + (size_t)m0 * Kdim + k0, ldsOff,
                          /*row_units(8B)=*/16, /*rows=*/16,
                          /*stride_units=*/(unsigned)(Kdim >> 2));
            __builtin_amdgcn_s_wait_tensorcnt(0);
        }
#else
        *(v8h*)&sA[arow][acol] = *(const v8h*)(A + (size_t)(m0 + arow) * Kdim + k0 + acol);
#endif
        if (ks + 1 < ksteps)
            __builtin_prefetch(Bw + (size_t)bn * Kdim + k0 + 64 + bk, 0, 1);
        __syncthreads();

        union { v8h h8[2]; v16h v; } a0, a1;
        a0.h8[0] = *(const v8h*)&sA[fr][fk];
        a0.h8[1] = *(const v8h*)&sA[fr][fk + 8];
        a1.h8[0] = *(const v8h*)&sA[fr][32 + fk];
        a1.h8[1] = *(const v8h*)&sA[fr][32 + fk + 8];
        v16h b0 = *(const v16h*)(Bw + (size_t)bn * Kdim + k0 + bk);
        v16h b1 = *(const v16h*)(Bw + (size_t)bn * Kdim + k0 + 32 + bk);

        acc = __builtin_amdgcn_wmma_f32_16x16x32_f16(
                  false, a0.v, false, b0, (short)0, acc, false, false);
        acc = __builtin_amdgcn_wmma_f32_16x16x32_f16(
                  false, a1.v, false, b1, (short)0, acc, false, false);
        __syncthreads();
    }

    const long long obase = (long long)bb * sO_b + (long long)ff * sO_f;
    const int g8 = (lane >> 4) * 8;
    const int nl = lane & 15;

    if (mode == MODE_ADDF32) {
        float* o = (float*)Out + obase;
        #pragma unroll
        for (int v = 0; v < 8; ++v)
            o[(size_t)(m0 + g8 + v) * ld_out + ntile + nl] += acc[v];
    } else if (mode == MODE_ZT) {
        // transposed f16 store: zT[b][channel][t]
        int b  = m0 / Tdim;
        int t0 = (m0 - b * Tdim) + g8;
        _Float16* o = (_Float16*)Out + ((size_t)b * Nout + ntile + nl) * Tdim + t0;
        v8h r;
        #pragma unroll
        for (int v = 0; v < 8; ++v) r[v] = (_Float16)acc[v];
        *(v8h*)o = r;
    } else {
        _Float16* o = (_Float16*)Out + obase;
        #pragma unroll
        for (int v = 0; v < 8; ++v) {
            float x = acc[v];
            if (mode == MODE_GELU) x = gelu_exact(x);
            o[(size_t)(m0 + g8 + v) * ld_out + ntile + nl] = (_Float16)x;
        }
    }
}

// ---------------- readout ----------------
__global__ void k_readout(const float* __restrict__ h, const float* __restrict__ wr,
                          float* __restrict__ out) {
    int i = blockIdx.x * blockDim.x + threadIdx.x;
    if (i >= BATCH * KQ) return;
    int b = i / KQ, q = i % KQ;
    const float* hp = h + ((size_t)b * TLEN + PTRN + q) * NCH;
    float s = 0.0f;
    #pragma unroll 16
    for (int n = 0; n < NCH; ++n) s += hp[n] * wr[n];
    out[i] = s;
}

// ---------------- launcher ----------------
extern "C" void kernel_launch(void* const* d_in, const int* in_sizes, int n_in,
                              void* d_out, int out_size, void* d_ws, size_t ws_size,
                              hipStream_t stream) {
    (void)in_sizes; (void)n_in; (void)out_size; (void)ws_size;
    const float* X_train  = (const float*)d_in[0];
    const float* y_train  = (const float*)d_in[1];
    const float* X_query  = (const float*)d_in[2];
    const float* W_in     = (const float*)d_in[3];
    const float* W_ro     = (const float*)d_in[4];
    const float* phi      = (const float*)d_in[5];
    const float* W_branch = (const float*)d_in[6];
    const float* full_mix = (const float*)d_in[7];
    const float* W_out    = (const float*)d_in[8];
    const float* fc1_in   = (const float*)d_in[9];
    const float* fc2_in   = (const float*)d_in[10];
    const float* fc1_out  = (const float*)d_in[11];
    const float* fc2_out  = (const float*)d_in[12];
    float* out = (float*)d_out;

    char* base = (char*)d_ws;
    size_t off = 0;
    auto carve = [&](size_t bytes) -> void* {
        void* p = base + off;
        off = (off + bytes + 255) & ~(size_t)255;
        return p;
    };
    _Float16* toep16  = (_Float16*)carve((size_t)RFILT * TLEN * TLEN * 2);
    _Float16* fc1i16  = (_Float16*)carve((size_t)NLAY * HID * NCH * 2);
    _Float16* fc2i16  = (_Float16*)carve((size_t)NLAY * NCH * HID * 2);
    _Float16* fc1o16  = (_Float16*)carve((size_t)NLAY * HID * NCH * 2);
    _Float16* fc2o16  = (_Float16*)carve((size_t)NLAY * NCH * HID * 2);
    _Float16* wb16    = (_Float16*)carve((size_t)NLAY * NCH * NCH * 2);
    _Float16* wo16    = (_Float16*)carve((size_t)NLAY * NCH * NCH * 2);
    _Float16* mixT16  = (_Float16*)carve((size_t)NLAY * NCH * RFILT * NCH * 2);
    float*    h       = (float*)   carve((size_t)MROWS * NCH * 4);
    _Float16* hn16    = (_Float16*)carve((size_t)MROWS * NCH * 2);
    _Float16* mid16   = (_Float16*)carve((size_t)MROWS * HID * 2);
    _Float16* zT16    = (_Float16*)carve((size_t)BATCH * NCH * TLEN * 2);
    _Float16* basis16 = (_Float16*)carve((size_t)MROWS * RFILT * NCH * 2);
    _Float16* y16     = (_Float16*)carve((size_t)MROWS * NCH * 2);

    auto f2h = [&](const float* s, _Float16* d, int n) {
        k_f2h<<<dim3((n + 255) / 256), dim3(256), 0, stream>>>(s, d, n);
    };
    // --- weight prep (cheap, once per call) ---
    f2h(fc1_in,  fc1i16, NLAY * HID * NCH);
    f2h(fc2_in,  fc2i16, NLAY * NCH * HID);
    f2h(fc1_out, fc1o16, NLAY * HID * NCH);
    f2h(fc2_out, fc2o16, NLAY * NCH * HID);
    f2h(W_branch, wb16,  NLAY * NCH * NCH);
    f2h(W_out,    wo16,  NLAY * NCH * NCH);
    k_mixT<<<dim3(256), dim3(256), 0, stream>>>(full_mix, mixT16, NLAY * NCH * RFILT * NCH);
    k_toep<<<dim3(4096), dim3(256), 0, stream>>>(phi, toep16, RFILT * TLEN * TLEN);

    // --- embed ---
    k_embed<<<dim3((MROWS * NCH + 255) / 256), dim3(256), 0, stream>>>(
        X_train, y_train, X_query, W_in, h, MROWS * NCH);

    auto gemm = [&](const _Float16* A, const _Float16* Bw, void* Out,
                    int M, int Nout, int Kdim, int mode, int ld_out,
                    int causal, int nz, int nf,
                    long long sAf, long long sBb, long long sOb, long long sOf) {
        dim3 g(M / 16, Nout / 64, nz);
        k_gemm<<<g, dim3(128), 0, stream>>>(A, Bw, Out, Nout, Kdim,
                                            sAf, sBb, sOb, sOf,
                                            nf, ld_out, mode, causal, TLEN);
    };

    for (int l = 0; l < NLAY; ++l) {
        const _Float16* fc1i = fc1i16 + (size_t)l * HID * NCH;
        const _Float16* fc2i = fc2i16 + (size_t)l * NCH * HID;
        const _Float16* fc1o = fc1o16 + (size_t)l * HID * NCH;
        const _Float16* fc2o = fc2o16 + (size_t)l * NCH * HID;
        const _Float16* wbl  = wb16   + (size_t)l * NCH * NCH;
        const _Float16* wol  = wo16   + (size_t)l * NCH * NCH;
        const _Float16* mixl = mixT16 + (size_t)l * NCH * RFILT * NCH;

        // h += mlp_in(rmsnorm(h))
        k_rmsnorm<<<dim3(MROWS / 4), dim3(128), 0, stream>>>(h, hn16);
        gemm(hn16, fc1i, mid16, MROWS, HID, NCH, MODE_GELU, HID, 0, 1, 1, 0, 0, 0, 0);
        gemm(mid16, fc2i, h,    MROWS, NCH, HID, MODE_ADDF32, NCH, 0, 1, 1, 0, 0, 0, 0);

        // z = rmsnorm(h) @ W_branch^T, stored transposed per batch: zT[b][c][t]
        k_rmsnorm<<<dim3(MROWS / 4), dim3(128), 0, stream>>>(h, hn16);
        gemm(hn16, wbl, zT16, MROWS, NCH, NCH, MODE_ZT, 0, 0, 1, 1, 0, 0, 0, 0);

        // basis[b][t][f][c] : causal Toeplitz GEMM, batched over (b,f)
        gemm(toep16, zT16, basis16, TLEN, NCH, TLEN, MODE_F16, RFILT * NCH,
             /*causal=*/1, /*nz=*/BATCH * RFILT, /*nf=*/RFILT,
             (long long)TLEN * TLEN,          // A per-filter stride
             (long long)NCH * TLEN,           // B per-batch stride
             (long long)TLEN * RFILT * NCH,   // Out per-batch stride
             (long long)NCH);                 // Out per-filter stride

        // y = basis @ mixT^T ; h += y @ W_out^T
        gemm(basis16, mixl, y16, MROWS, NCH, RFILT * NCH, MODE_F16, NCH, 0, 1, 1, 0, 0, 0, 0);
        gemm(y16, wol, h, MROWS, NCH, NCH, MODE_ADDF32, NCH, 0, 1, 1, 0, 0, 0, 0);

        // h += mlp_out(rmsnorm(h))
        k_rmsnorm<<<dim3(MROWS / 4), dim3(128), 0, stream>>>(h, hn16);
        gemm(hn16, fc1o, mid16, MROWS, HID, NCH, MODE_GELU, HID, 0, 1, 1, 0, 0, 0, 0);
        gemm(mid16, fc2o, h,    MROWS, NCH, HID, MODE_ADDF32, NCH, 0, 1, 1, 0, 0, 0, 0);
    }

    k_readout<<<dim3((BATCH * KQ + 127) / 128), dim3(128), 0, stream>>>(h, W_ro, out);
}
